// MultiHeadAttention_39943195852936
// MI455X (gfx1250) — matrix operand order repaired and checked by reading, not compiled, not run on hardware
//
#include <hip/hip_runtime.h>

// MI455X / gfx1250: wave32, WMMA 16x16x32 bf16->f32.
// Pipeline: cvt f32->bf16, fused QKV GEMM (+per-head LayerNorm epilogue),
// flash-attention (online softmax, all matmuls via v_wmma), output GEMM.
// GEMM: 128x128 tile, BK=64, double-buffered LDS, async global->LDS staging.

typedef __bf16 bf16;
typedef bf16  v16bf  __attribute__((ext_vector_type(16)));
typedef bf16  bf16x8 __attribute__((ext_vector_type(8)));
typedef bf16  bf16x4 __attribute__((ext_vector_type(4)));
typedef float v8f    __attribute__((ext_vector_type(8)));
typedef int   v4i    __attribute__((vector_size(4 * sizeof(int))));  // matches builtin

#define BB 2
#define SS 2048
#define DD 1024
#define HH 16
#define HD 64
#define BM 128
#define BN 128
#define BK 64

// ---- async global->LDS staging (CDNA5 GLOBAL_LOAD_ASYNC_TO_LDS_B128) ------
#if __has_builtin(__builtin_amdgcn_global_load_async_to_lds_b128)
#define HAVE_ASYNC_LDS 1
#else
#define HAVE_ASYNC_LDS 0
#endif

__device__ __forceinline__ void async_copy16(void* l, const void* g) {
#if HAVE_ASYNC_LDS
  __builtin_amdgcn_global_load_async_to_lds_b128(
      (__attribute__((address_space(1))) v4i*)(g),
      (__attribute__((address_space(3))) v4i*)(l), 0, 0);
#else
  *(int4*)l = *(const int4*)g;
#endif
}
__device__ __forceinline__ void wait_async_all() {
  asm volatile("s_wait_asynccnt 0" ::: "memory");
}

// ---- WMMA fragment helpers (contiguous k = hi*16+i order for A and B, wave32) -------------------
// A fragment, 16x32 (MxK): lane L: M=L&15, hi=L>>4;
//   e[i] = A[M][hi*16+i]
__device__ __forceinline__ v16bf frag_A32(const bf16* row, int hi) {
  bf16x8 lo = *(const bf16x8*)(row + hi * 16);
  bf16x8 up = *(const bf16x8*)(row + hi * 16 + 8);
  return __builtin_shufflevector(lo, up, 0,1,2,3,4,5,6,7,8,9,10,11,12,13,14,15);
}
// B fragment, 32x16 (KxN) from transposed tile Bt[N][K]: e[i] = Bt[N][hi*16+i]
__device__ __forceinline__ v16bf frag_B16(const bf16* p) {
  bf16x8 lo = *(const bf16x8*)(p);
  bf16x8 up = *(const bf16x8*)(p + 8);
  return __builtin_shufflevector(lo, up, 0,1,2,3,4,5,6,7,8,9,10,11,12,13,14,15);
}
__device__ __forceinline__ v8f wmma_bf16(v16bf a, v16bf b, v8f c) {
  return __builtin_amdgcn_wmma_f32_16x16x32_bf16(
      false, a, false, b, (short)0, c, false, false);
}

// ---- f32 -> bf16 conversion (vectorized x4) -------------------------------
__global__ __launch_bounds__(256) void cvt_f32_bf16_kernel(
    const float* __restrict__ s, bf16* __restrict__ d, int n4) {
  int i = blockIdx.x * 256 + threadIdx.x;
  if (i < n4) {
    const float4 v = ((const float4*)s)[i];
    bf16x4 o = { (bf16)v.x, (bf16)v.y, (bf16)v.z, (bf16)v.w };
    ((bf16x4*)d)[i] = o;
  }
}

// ---- GEMM: C[4096,1024] = X * Wt^T + bias ---------------------------------
// Wt is torch-layout [out,in] bf16 == transposed-B tile (Bt[n][k]=W[n][k]).
// 128(M) x 128(N) per WG, BK=64, double-buffered LDS, 1 barrier per K-step.
// 8 waves as 4(M) x 2(N); wave tile 32x64 -> 8 accumulators, 16 wmma/K-step.
// mode 0: +bias, per-head LayerNorm, store bf16 [B,H,S,HD]   (Q and K)
// mode 2: +bias, store bf16 transposed [B,H,HD,S]            (V)
// mode 3: +bias, store f32 row-major [M,N]                   (output proj)
__global__ __launch_bounds__(256) void gemm_kernel(
    const bf16* __restrict__ X, const bf16* __restrict__ Wt,
    const float* __restrict__ bias,
    const float* __restrict__ lng, const float* __restrict__ lnb,
    bf16* __restrict__ out_bf, float* __restrict__ out_f, int mode) {
  __shared__ union {
    struct {
      __align__(16) bf16 X[2][BM][BK];
      __align__(16) bf16 W[2][BN][BK];
    } t;                      // 64 KB (double-buffered tiles)
    float C[BM][BN];          // 64 KB (LN epilogue scratch)
  } sm;

  const int tid  = threadIdx.x;
  const int lane = tid & 31, wid = tid >> 5;
  const int l15  = lane & 15, hi = lane >> 4;
  const int waveM = wid >> 1, waveN = wid & 1;
  const int tM = blockIdx.x & 31, tN = blockIdx.x >> 5;  // 32 x 8 tiles
  const int m0 = tM * BM, n0 = tN * BN;
  const int crow = tid >> 1, cseg = (tid & 1) * 32;      // copy: 1 row-half/thread

  const bf16* xsrc = X  + (size_t)(m0 + crow) * DD + cseg;
  const bf16* wsrc = Wt + (size_t)(n0 + crow) * DD + cseg;

  // prefetch K-block 0 into buffer 0
#pragma unroll
  for (int q = 0; q < 4; ++q) {
    async_copy16(&sm.t.X[0][crow][cseg + q * 8], xsrc + q * 8);
    async_copy16(&sm.t.W[0][crow][cseg + q * 8], wsrc + q * 8);
  }

  v8f acc[2][4];
#pragma unroll
  for (int mt = 0; mt < 2; ++mt)
#pragma unroll
    for (int nt = 0; nt < 4; ++nt) acc[mt][nt] = {};

  for (int it = 0; it < DD / BK; ++it) {
    const int buf = it & 1;
    wait_async_all();
    __syncthreads();                    // buffer `buf` ready; `buf^1` free
    if (it + 1 < DD / BK) {             // stream next K-block into other buffer
      const bf16* xs = xsrc + (it + 1) * BK;
      const bf16* ws = wsrc + (it + 1) * BK;
#pragma unroll
      for (int q = 0; q < 4; ++q) {
        async_copy16(&sm.t.X[buf ^ 1][crow][cseg + q * 8], xs + q * 8);
        async_copy16(&sm.t.W[buf ^ 1][crow][cseg + q * 8], ws + q * 8);
      }
    }
#pragma unroll
    for (int ks = 0; ks < 2; ++ks) {
      v16bf a[2];
#pragma unroll
      for (int mt = 0; mt < 2; ++mt)
        a[mt] = frag_A32(&sm.t.X[buf][waveM * 32 + mt * 16 + l15][ks * 32], hi);
#pragma unroll
      for (int nt = 0; nt < 4; ++nt) {
        v16bf b = frag_B16(&sm.t.W[buf][waveN * 64 + nt * 16 + l15][ks * 32 + hi * 16]);
#pragma unroll
        for (int mt = 0; mt < 2; ++mt)
          acc[mt][nt] = wmma_bf16(a[mt], b, acc[mt][nt]);
      }
    }
  }
  __syncthreads();

  if (mode == 3) {  // final projection -> f32 out
#pragma unroll
    for (int mt = 0; mt < 2; ++mt)
#pragma unroll
      for (int nt = 0; nt < 4; ++nt)
#pragma unroll
        for (int j = 0; j < 8; ++j) {
          int mr = m0 + waveM * 32 + mt * 16 + hi * 8 + j;
          int nc = n0 + waveN * 64 + nt * 16 + l15;
          out_f[(size_t)mr * DD + nc] = acc[mt][nt][j] + bias[nc];
        }
    return;
  }
  if (mode == 2) {  // V -> bf16 [B,H,HD,S]
#pragma unroll
    for (int mt = 0; mt < 2; ++mt)
#pragma unroll
      for (int nt = 0; nt < 4; ++nt)
#pragma unroll
        for (int j = 0; j < 8; ++j) {
          int mr = m0 + waveM * 32 + mt * 16 + hi * 8 + j;
          int nc = n0 + waveN * 64 + nt * 16 + l15;
          int b = mr >> 11, s = mr & (SS - 1);
          out_bf[(((size_t)b * HH + (nc >> 6)) * HD + (nc & (HD - 1))) * SS + s] =
              (bf16)(acc[mt][nt][j] + bias[nc]);
        }
    return;
  }
  // mode 0: per-head LayerNorm epilogue; N-tile = 2 heads, 256 threads =
  // 128 rows x 2 heads -> one (row,head) per thread, no cross-thread reduce.
#pragma unroll
  for (int mt = 0; mt < 2; ++mt)
#pragma unroll
    for (int nt = 0; nt < 4; ++nt)
#pragma unroll
      for (int j = 0; j < 8; ++j) {
        int r = waveM * 32 + mt * 16 + hi * 8 + j;
        int c = waveN * 64 + nt * 16 + l15;
        sm.C[r][c] = acc[mt][nt][j] + bias[n0 + c];
      }
  __syncthreads();
  {
    const int r = tid >> 1, hd0 = (tid & 1) * HD;
    float s1 = 0.f, s2 = 0.f;
#pragma unroll 8
    for (int i = 0; i < HD; ++i) {
      float v = sm.C[r][hd0 + i];
      s1 += v;
      s2 += v * v;
    }
    float mu   = s1 * (1.f / HD);
    float var  = s2 * (1.f / HD) - mu * mu;
    float rstd = rsqrtf(var + 1e-5f);
    int mr = m0 + r;
    int b = mr >> 11, s = mr & (SS - 1);
    int h = (n0 >> 6) + (tid & 1);
    bf16* orow = out_bf + (((size_t)b * HH + h) * SS + s) * HD;
#pragma unroll 8
    for (int i = 0; i < HD; ++i) {
      float y = (sm.C[r][hd0 + i] - mu) * rstd * lng[i] + lnb[i];
      orow[i] = (bf16)y;
    }
  }
}

// ---- Flash attention: per (b,h), 128 query rows per WG --------------------
// Q,K bf16 [B,H,S,64]; Vt bf16 [B,H,64,S]; O bf16 [B,S,D].
// 8 waves x 16 query rows; online softmax in f32; both matmuls via WMMA.
__global__ __launch_bounds__(256) void flash_attn_kernel(
    const bf16* __restrict__ Q, const bf16* __restrict__ K,
    const bf16* __restrict__ Vt, bf16* __restrict__ O) {
  __shared__ __align__(16) bf16 Ks[64][64];     // Bt tile for Q*K^T (keys x hd)
  __shared__ __align__(16) bf16 Vs[64][64];     // Bt tile for P*V  (hd x keys)
  __shared__ __align__(16) bf16 Ps[8][16][64];  // per-wave P relayout scratch

  const int tid  = threadIdx.x;
  const int lane = tid & 31, wid = tid >> 5;
  const int l15  = lane & 15, hi = lane >> 4;
  const int qblk = blockIdx.x & 15, bh = blockIdx.x >> 4;
  const int lrow = tid >> 2, lseg = tid & 3;

  // Q A-fragments stay in registers for the whole KV sweep
  const int   qrow = qblk * 128 + wid * 16 + l15;
  const bf16* qp   = Q + ((size_t)bh * SS + qrow) * HD;
  v16bf aq0 = frag_A32(qp, hi);       // hd 0..31
  v16bf aq1 = frag_A32(qp + 32, hi);  // hd 32..63

  v8f o[4];
  o[0] = {}; o[1] = {}; o[2] = {}; o[3] = {};
  float mrow[8], lsum[8];
#pragma unroll
  for (int j = 0; j < 8; ++j) { mrow[j] = -1e30f; lsum[j] = 0.f; }

  const bf16* kbase = K  + (size_t)bh * SS * HD;
  const bf16* vbase = Vt + (size_t)bh * HD * SS;

  for (int j0 = 0; j0 < SS; j0 += 64) {
    __syncthreads();
    {  // stage K block (64 keys x 64 hd) and Vt block (64 hd x 64 keys)
      const bf16* ksrc = kbase + (size_t)(j0 + lrow) * HD + lseg * 16;
      async_copy16(&Ks[lrow][lseg * 16], ksrc);
      async_copy16(&Ks[lrow][lseg * 16 + 8], ksrc + 8);
      const bf16* vsrc = vbase + (size_t)lrow * SS + j0 + lseg * 16;
      async_copy16(&Vs[lrow][lseg * 16], vsrc);
      async_copy16(&Vs[lrow][lseg * 16 + 8], vsrc + 8);
    }
    wait_async_all();
    __syncthreads();

    // S = Q * K^T  (16 x 64 per wave)
    v8f s[4];
    s[0] = {}; s[1] = {}; s[2] = {}; s[3] = {};
#pragma unroll
    for (int ks = 0; ks < 2; ++ks) {
      v16bf a = ks ? aq1 : aq0;
#pragma unroll
      for (int nt = 0; nt < 4; ++nt) {
        v16bf b = frag_B16(&Ks[nt * 16 + l15][ks * 32 + hi * 16]);
        s[nt] = wmma_bf16(a, b, s[nt]);
      }
    }
#pragma unroll
    for (int nt = 0; nt < 4; ++nt) s[nt] *= 0.125f;  // 1/sqrt(64)

    // online softmax; C-layout row = hi*8+j lives on the 16 lanes of a half
#pragma unroll
    for (int j = 0; j < 8; ++j) {
      float v = fmaxf(fmaxf(s[0][j], s[1][j]), fmaxf(s[2][j], s[3][j]));
#pragma unroll
      for (int msk = 1; msk < 16; msk <<= 1) v = fmaxf(v, __shfl_xor(v, msk, 32));
      float mn  = fmaxf(mrow[j], v);
      float cor = __expf(mrow[j] - mn);
      mrow[j] = mn;
      float rsum = 0.f;
#pragma unroll
      for (int nt = 0; nt < 4; ++nt) {
        float p = __expf(s[nt][j] - mn);
        s[nt][j] = p;
        rsum += p;
      }
#pragma unroll
      for (int msk = 1; msk < 16; msk <<= 1) rsum += __shfl_xor(rsum, msk, 32);
      lsum[j] = lsum[j] * cor + rsum;
#pragma unroll
      for (int nt = 0; nt < 4; ++nt) o[nt][j] *= cor;
    }

    // relayout P: C-fragment -> bf16 A-fragment via wave-private LDS
#pragma unroll
    for (int nt = 0; nt < 4; ++nt)
#pragma unroll
      for (int j = 0; j < 8; ++j)
        Ps[wid][hi * 8 + j][nt * 16 + l15] = (bf16)s[nt][j];
    asm volatile("s_wait_dscnt 0" ::: "memory");  // same-wave LDS RAW fence

    // O += P * V
#pragma unroll
    for (int ks = 0; ks < 2; ++ks) {
      v16bf ap = frag_A32(&Ps[wid][l15][ks * 32], hi);
#pragma unroll
      for (int nt = 0; nt < 4; ++nt) {
        v16bf bv = frag_B16(&Vs[nt * 16 + l15][ks * 32 + hi * 16]);
        o[nt] = wmma_bf16(ap, bv, o[nt]);
      }
    }
  }

  // epilogue: normalize by l and store bf16 [B,S,D]
  const int b = bh >> 4, h = bh & 15;
  float rl[8];
#pragma unroll
  for (int j = 0; j < 8; ++j) rl[j] = 1.f / lsum[j];
#pragma unroll
  for (int nt = 0; nt < 4; ++nt)
#pragma unroll
    for (int j = 0; j < 8; ++j) {
      int q = qblk * 128 + wid * 16 + hi * 8 + j;
      int d = nt * 16 + l15;
      O[((size_t)b * SS + q) * DD + h * HD + d] = (bf16)(o[nt][j] * rl[j]);
    }
}

// ---- host-side orchestration ----------------------------------------------
extern "C" void kernel_launch(void* const* d_in, const int* in_sizes, int n_in,
                              void* d_out, int out_size, void* d_ws, size_t ws_size,
                              hipStream_t stream) {
  (void)in_sizes; (void)n_in; (void)out_size; (void)ws_size;
  const float* query  = (const float*)d_in[0];
  const float* q_w    = (const float*)d_in[1];
  const float* q_b    = (const float*)d_in[2];
  const float* k_w    = (const float*)d_in[3];
  const float* k_b    = (const float*)d_in[4];
  const float* v_w    = (const float*)d_in[5];
  const float* v_b    = (const float*)d_in[6];
  const float* o_w    = (const float*)d_in[7];
  const float* o_b    = (const float*)d_in[8];
  const float* q_ln_g = (const float*)d_in[9];
  const float* q_ln_b = (const float*)d_in[10];
  const float* k_ln_g = (const float*)d_in[11];
  const float* k_ln_b = (const float*)d_in[12];
  float* out = (float*)d_out;

  char* w = (char*)d_ws;  // 48 MB workspace layout
  bf16* xb  = (bf16*)(w + (size_t)0);           // X bf16          [4096,1024]  8MB
  bf16* wqb = (bf16*)(w + ((size_t)8  << 20));  // q_w bf16        [1024,1024]  2MB
  bf16* wkb = (bf16*)(w + ((size_t)10 << 20));
  bf16* wvb = (bf16*)(w + ((size_t)12 << 20));
  bf16* wob = (bf16*)(w + ((size_t)14 << 20));
  bf16* qh  = (bf16*)(w + ((size_t)16 << 20));  // Q post-LN       [B,H,S,64]   8MB
  bf16* kh  = (bf16*)(w + ((size_t)24 << 20));  // K post-LN       [B,H,S,64]   8MB
  bf16* vt  = (bf16*)(w + ((size_t)32 << 20));  // V transposed    [B,H,64,S]   8MB
  bf16* at  = (bf16*)(w + ((size_t)40 << 20));  // attention out   [B,S,D]      8MB

  const int nX4 = BB * SS * DD / 4;  // 1,048,576
  const int nW4 = DD * DD / 4;       //   262,144
  cvt_f32_bf16_kernel<<<nX4 / 256, 256, 0, stream>>>(query, xb, nX4);
  cvt_f32_bf16_kernel<<<nW4 / 256, 256, 0, stream>>>(q_w, wqb, nW4);
  cvt_f32_bf16_kernel<<<nW4 / 256, 256, 0, stream>>>(k_w, wkb, nW4);
  cvt_f32_bf16_kernel<<<nW4 / 256, 256, 0, stream>>>(v_w, wvb, nW4);
  cvt_f32_bf16_kernel<<<nW4 / 256, 256, 0, stream>>>(o_w, wob, nW4);

  const int gemmGrid = (BB * SS / BM) * (DD / BN);  // 32 * 8 = 256 WGs
  gemm_kernel<<<gemmGrid, 256, 0, stream>>>(xb, wqb, q_b, q_ln_g, q_ln_b, qh, nullptr, 0);
  gemm_kernel<<<gemmGrid, 256, 0, stream>>>(xb, wkb, k_b, k_ln_g, k_ln_b, kh, nullptr, 0);
  gemm_kernel<<<gemmGrid, 256, 0, stream>>>(xb, wvb, v_b, nullptr, nullptr, vt, nullptr, 2);

  flash_attn_kernel<<<BB * HH * (SS / 128), 256, 0, stream>>>(qh, kh, vt, at);

  gemm_kernel<<<gemmGrid, 256, 0, stream>>>(at, wob, o_b, nullptr, nullptr, nullptr, out, 3);
}
